// SequentialPhysicsInformedGNN_41051297415283
// MI455X (gfx1250) — compile-verified, matrix operand order checked by reading, not run on hardware
//
#include <hip/hip_runtime.h>
#include <hip/hip_bf16.h>

// ---------------------------------------------------------------------------
// CDNA5 (gfx1250) WMMA implementation of SequentialPhysicsInformedGNN.
// Edge MLP runs on v_wmma_f32_16x16x32_f16; one wave32 owns a 16-edge tile.
// ---------------------------------------------------------------------------

typedef __attribute__((ext_vector_type(16))) _Float16 v16h;
typedef __attribute__((ext_vector_type(8)))  float    v8f;

#define WAVES 8          // waves per block (256 threads)
#define HIDC  64         // hidden channels
#define HEADS 4
#define KPMAX 160        // max padded K (136 -> 160)

__device__ __forceinline__ float gelu_f(float x) {
    return 0.5f * x * (1.0f + erff(x * 0.70710678118654752f));
}

// order-preserving float->uint key for atomicMax-based segment max
__device__ __forceinline__ unsigned enc_f(float f) {
    unsigned b = __float_as_uint(f);
    return (b & 0x80000000u) ? ~b : (b | 0x80000000u);
}
__device__ __forceinline__ float dec_f(unsigned k) {
    unsigned b = (k & 0x80000000u) ? (k & 0x7FFFFFFFu) : ~k;
    return __uint_as_float(b);
}

// Load a 16x32 f16 A-fragment for this lane from a row-major LDS row.
// ISA layout: lane (row = lane&15, half = lane>>4) holds K values
//   e<8 : K = 8*half + e        e>=8 : K = 8*half + 8 + e
__device__ __forceinline__ v16h load_a_frag(const _Float16* row, int kbase, int half_) {
    v16h a;
#pragma unroll
    for (int e = 0; e < 16; ++e) {
        int k = (e < 8) ? (8 * half_ + e) : (8 * half_ + 8 + e);
        a[e] = row[kbase + k];
    }
    return a;
}

// ---------------------------------------------------------------------------
// Weight conversion: f32 [K][N] -> f16 fragment-ordered [kt*ntl+nt][lane][e]
// (B-fragment layout mirrors A: lane holds col n = nt*16 + (lane&15),
//  K values per e/half as above). Zero-padded beyond K/N.
// ---------------------------------------------------------------------------
__global__ void cvt_w(const float* __restrict__ W, int K, int Nin,
                      int nkt, int ntl, _Float16* __restrict__ out) {
    int idx = blockIdx.x * 256 + threadIdx.x;
    int total = nkt * ntl * 32 * 16;
    if (idx >= total) return;
    int e     = idx & 15;
    int lane  = (idx >> 4) & 31;
    int f     = idx >> 9;
    int nt    = f % ntl;
    int kt    = f / ntl;
    int half_ = lane >> 4;
    int n     = nt * 16 + (lane & 15);
    int k     = kt * 32 + ((e < 8) ? (8 * half_ + e) : (8 * half_ + 8 + e));
    float v = (k < K && n < Nin) ? W[(long)k * Nin + n] : 0.0f;
    out[idx] = (_Float16)v;
}

// ---------------------------------------------------------------------------
// Per-layer init: aggr=0, segment sums=0, segment max keys=enc(-inf)
// ---------------------------------------------------------------------------
__global__ void init_layer(float* __restrict__ xnext, float* __restrict__ nodesum,
                           unsigned* __restrict__ nodemax, int N_) {
    int i = blockIdx.x * 256 + threadIdx.x;
    if (i < N_ * HIDC) xnext[i] = 0.0f;
    if (i < N_ * HEADS) { nodesum[i] = 0.0f; nodemax[i] = 0x007FFFFFu; /* enc(-inf) */ }
}

// ---------------------------------------------------------------------------
// Edge MLP: h = GELU chain of 3 GEMMs, logits = leakyrelu(h@Wa+ba).
// One wave per 16-edge tile; WMMA f32_16x16x32_f16 everywhere.
// ---------------------------------------------------------------------------
template <int NKT>
__global__ __launch_bounds__(256) void edge_mlp(
    const float* __restrict__ xin, int d, int d_in,
    const int* __restrict__ src, const int* __restrict__ dst,
    const float* __restrict__ eattr, int E_,
    const _Float16* __restrict__ Wf1, const float* __restrict__ b1,
    const _Float16* __restrict__ Wf2, const float* __restrict__ b2,
    const _Float16* __restrict__ Wf3, const float* __restrict__ b3,
    const _Float16* __restrict__ Wfa, const float* __restrict__ ba,
    _Float16* __restrict__ hbuf, float* __restrict__ logit,
    unsigned* __restrict__ nodemax)
{
    constexpr int KP = NKT * 32;
    __shared__ _Float16 ldsA[WAVES][16][KPMAX];  // m_in tile / h2 tile
    __shared__ _Float16 ldsH[WAVES][16][HIDC];   // h1 / h3 tile
    __shared__ int      ldsDst[WAVES][16];

    const int tid   = threadIdx.x;
    const int wave  = tid >> 5;
    const int lane  = tid & 31;
    const int half_ = lane >> 4;
    const int r     = lane & 15;      // edge row within tile (for A / output rows)
    const int col   = lane & 15;      // output column within 16-wide N tile
    const long tile = (long)blockIdx.x * WAVES + wave;
    const long e0   = tile * 16;

    // ---- phase 0: gather m_in = [x[dst] | x[src] | edge_attr] into LDS (f16)
    {
        long eg = e0 + r;
        bool ok = eg < E_;
        int di = ok ? dst[eg] : 0;
        int si = ok ? src[eg] : 0;
        if (half_ == 0) ldsDst[wave][r] = di;
        for (int c = half_; c < KP; c += 2) {
            float v = 0.0f;
            if (ok) {
                if (c < d)            v = xin[(long)di * d + c];
                else if (c < 2 * d)   v = xin[(long)si * d + (c - d)];
                else if (c < d_in)    v = eattr[eg * 8 + (c - 2 * d)];
            }
            ldsA[wave][r][c] = (_Float16)v;
        }
    }
    __syncthreads();

    const v16h* Bf1 = (const v16h*)Wf1;
    const v16h* Bf2 = (const v16h*)Wf2;
    const v16h* Bf3 = (const v16h*)Wf3;
    const v16h* Bfa = (const v16h*)Wfa;

    // ---- GEMM1: h1 = gelu(m_in @ W1 + b1)   [16 x KP] @ [KP x 64]
    {
        v16h afr[NKT];
#pragma unroll
        for (int kt = 0; kt < NKT; ++kt)
            afr[kt] = load_a_frag(&ldsA[wave][r][0], kt * 32, half_);
#pragma unroll
        for (int nt = 0; nt < 4; ++nt) {
            v8f acc = {};
#pragma unroll
            for (int kt = 0; kt < NKT; ++kt)
                acc = __builtin_amdgcn_wmma_f32_16x16x32_f16(
                    false, afr[kt], false, Bf1[(kt * 4 + nt) * 32 + lane],
                    (short)0, acc, false, false);
            float bias = b1[nt * 16 + col];
#pragma unroll
            for (int v = 0; v < 8; ++v) {
                int m = v + 8 * half_;
                ldsH[wave][m][nt * 16 + col] = (_Float16)gelu_f(acc[v] + bias);
            }
        }
    }
    __syncthreads();

    // ---- GEMM2: h2 = gelu(h1 @ W2 + b2)   [16 x 64] @ [64 x 64]  -> ldsA
    {
        v16h afr[2];
#pragma unroll
        for (int kt = 0; kt < 2; ++kt)
            afr[kt] = load_a_frag(&ldsH[wave][r][0], kt * 32, half_);
#pragma unroll
        for (int nt = 0; nt < 4; ++nt) {
            v8f acc = {};
#pragma unroll
            for (int kt = 0; kt < 2; ++kt)
                acc = __builtin_amdgcn_wmma_f32_16x16x32_f16(
                    false, afr[kt], false, Bf2[(kt * 4 + nt) * 32 + lane],
                    (short)0, acc, false, false);
            float bias = b2[nt * 16 + col];
#pragma unroll
            for (int v = 0; v < 8; ++v) {
                int m = v + 8 * half_;
                ldsA[wave][m][nt * 16 + col] = (_Float16)gelu_f(acc[v] + bias);
            }
        }
    }
    __syncthreads();

    // ---- GEMM3: h3 = gelu(h2 @ W3 + b3)  -> ldsH
    {
        v16h afr[2];
#pragma unroll
        for (int kt = 0; kt < 2; ++kt)
            afr[kt] = load_a_frag(&ldsA[wave][r][0], kt * 32, half_);
#pragma unroll
        for (int nt = 0; nt < 4; ++nt) {
            v8f acc = {};
#pragma unroll
            for (int kt = 0; kt < 2; ++kt)
                acc = __builtin_amdgcn_wmma_f32_16x16x32_f16(
                    false, afr[kt], false, Bf3[(kt * 4 + nt) * 32 + lane],
                    (short)0, acc, false, false);
            float bias = b3[nt * 16 + col];
#pragma unroll
            for (int v = 0; v < 8; ++v) {
                int m = v + 8 * half_;
                ldsH[wave][m][nt * 16 + col] = (_Float16)gelu_f(acc[v] + bias);
            }
        }
    }
    __syncthreads();

    // ---- store h3 to global (f16), 32 channels per lane, 128-bit moves
    {
        long eg = e0 + r;
        if (eg < E_) {
            const _Float16* s = &ldsH[wave][r][half_ * 32];
            _Float16* g = &hbuf[eg * HIDC + half_ * 32];
#pragma unroll
            for (int i = 0; i < 4; ++i)
                ((float4*)g)[i] = ((const float4*)s)[i];
        }
    }

    // ---- GEMM4: logits = leakyrelu(h3 @ Wa + ba)   (Wa padded 64x16, 1 N-tile)
    {
        v16h afr[2];
#pragma unroll
        for (int kt = 0; kt < 2; ++kt)
            afr[kt] = load_a_frag(&ldsH[wave][r][0], kt * 32, half_);
        v8f acc = {};
#pragma unroll
        for (int kt = 0; kt < 2; ++kt)
            acc = __builtin_amdgcn_wmma_f32_16x16x32_f16(
                false, afr[kt], false, Bfa[kt * 32 + lane],
                (short)0, acc, false, false);
        if (col < HEADS) {
            float bias = ba[col];
#pragma unroll
            for (int v = 0; v < 8; ++v) {
                int m = v + 8 * half_;
                long eg = e0 + m;
                if (eg < E_) {
                    float z = acc[v] + bias;
                    float l = (z > 0.0f) ? z : 0.1f * z;
                    logit[eg * HEADS + col] = l;
                    int dd = ldsDst[wave][m];
                    atomicMax(&nodemax[(long)dd * HEADS + col], enc_f(l));
                }
            }
        }
    }
}

// ---------------------------------------------------------------------------
// Segment softmax pieces
// ---------------------------------------------------------------------------
__global__ void edge_exp(const int* __restrict__ dst, float* __restrict__ logit,
                         const unsigned* __restrict__ nodemax,
                         float* __restrict__ nodesum, long E4) {
    long i = (long)blockIdx.x * 256 + threadIdx.x;
    if (i >= E4) return;
    long e = i >> 2; int h = (int)(i & 3);
    int d_ = dst[e];
    float m = dec_f(nodemax[(long)d_ * HEADS + h]);
    float v = expf(logit[i] - m);
    logit[i] = v;   // reuse buffer: now holds exp values
    atomicAdd(&nodesum[(long)d_ * HEADS + h], v);
}

__global__ void edge_w(const int* __restrict__ dst, const float* __restrict__ ebuf,
                       const float* __restrict__ nodesum, float* __restrict__ wbuf,
                       long E_) {
    long e = (long)blockIdx.x * 256 + threadIdx.x;
    if (e >= E_) return;
    int d_ = dst[e];
    float w = 0.0f;
#pragma unroll
    for (int h = 0; h < HEADS; ++h)
        w += ebuf[e * HEADS + h] / (nodesum[(long)d_ * HEADS + h] + 1e-16f);
    wbuf[e] = 0.25f * w;
}

__global__ void edge_aggr(const int* __restrict__ dst, const float* __restrict__ wbuf,
                          const _Float16* __restrict__ hbuf, float* __restrict__ xnext,
                          long EC_) {
    long i = (long)blockIdx.x * 256 + threadIdx.x;
    if (i >= EC_) return;
    long e = i >> 6; int c = (int)(i & 63);
    atomicAdd(&xnext[(long)dst[e] * HIDC + c], wbuf[e] * (float)hbuf[i]);
}

__global__ void node_gelu(float* __restrict__ x, int n) {
    int i = blockIdx.x * 256 + threadIdx.x;
    if (i < n) x[i] = gelu_f(x[i]);
}

// ---------------------------------------------------------------------------
// Final per-node FC head: 64 -> 32 -> 16 -> 3 (tiny, plain VALU)
// ---------------------------------------------------------------------------
__global__ void fc_kernel(const float* __restrict__ xin,
                          const float* __restrict__ W1, const float* __restrict__ b1,
                          const float* __restrict__ W2, const float* __restrict__ b2,
                          const float* __restrict__ W3, const float* __restrict__ b3,
                          float* __restrict__ out, int N_) {
    int n = blockIdx.x * 256 + threadIdx.x;
    if (n >= N_) return;
    float h[64];
#pragma unroll
    for (int i = 0; i < 64; ++i) h[i] = xin[(long)n * 64 + i];
    float g1[32];
    for (int j = 0; j < 32; ++j) {
        float s = b1[j];
        for (int i = 0; i < 64; ++i) s += h[i] * W1[i * 32 + j];
        g1[j] = gelu_f(s);
    }
    float g2[16];
    for (int j = 0; j < 16; ++j) {
        float s = b2[j];
        for (int i = 0; i < 32; ++i) s += g1[i] * W2[i * 16 + j];
        g2[j] = gelu_f(s);
    }
    for (int k = 0; k < 3; ++k) {
        float s = b3[k];
        for (int i = 0; i < 16; ++i) s += g2[i] * W3[i * 3 + k];
        out[(long)n * 3 + k] = s;
    }
}

// ---------------------------------------------------------------------------
// Host orchestration
// ---------------------------------------------------------------------------
extern "C" void kernel_launch(void* const* d_in, const int* in_sizes, int n_in,
                              void* d_out, int out_size, void* d_ws, size_t ws_size,
                              hipStream_t stream) {
    (void)n_in; (void)out_size; (void)ws_size;
    const float* x     = (const float*)d_in[0];
    const int*   src   = (const int*)d_in[1];
    const int*   dst   = (const int*)d_in[2];
    const float* eattr = (const float*)d_in[3];
    const int  N_ = in_sizes[0] / 16;
    const long E_ = in_sizes[1];

    // gnn params: 4 + layer*8 + {W1,b1,W2,b2,W3,b3,Wa,ba}
    const float *gW[4][4], *gB[4][4];
    for (int l = 0; l < 4; ++l) {
        int base = 4 + l * 8;
        for (int m = 0; m < 4; ++m) {
            gW[l][m] = (const float*)d_in[base + 2 * m];
            gB[l][m] = (const float*)d_in[base + 2 * m + 1];
        }
    }
    const float* fcW1 = (const float*)d_in[36]; const float* fcb1 = (const float*)d_in[37];
    const float* fcW2 = (const float*)d_in[38]; const float* fcb2 = (const float*)d_in[39];
    const float* fcW3 = (const float*)d_in[40]; const float* fcb3 = (const float*)d_in[41];

    // workspace carve-up (256B aligned)
    char* ws = (char*)d_ws;
    size_t off = 0;
    auto alloc = [&](size_t bytes) -> void* {
        void* p = ws + off;
        off = (off + bytes + 255) & ~(size_t)255;
        return p;
    };

    _Float16* wf[4][4];
    int nkt1[4];
    for (int l = 0; l < 4; ++l) {
        nkt1[l] = (l == 0) ? 2 : 5;
        wf[l][0] = (_Float16*)alloc((size_t)nkt1[l] * 4 * 512 * sizeof(_Float16));
        wf[l][1] = (_Float16*)alloc((size_t)2 * 4 * 512 * sizeof(_Float16));
        wf[l][2] = (_Float16*)alloc((size_t)2 * 4 * 512 * sizeof(_Float16));
        wf[l][3] = (_Float16*)alloc((size_t)2 * 1 * 512 * sizeof(_Float16));
    }
    _Float16* hbuf    = (_Float16*)alloc((size_t)E_ * HIDC * sizeof(_Float16));
    float*    logit   = (float*)alloc((size_t)E_ * HEADS * sizeof(float));
    float*    wbuf    = (float*)alloc((size_t)E_ * sizeof(float));
    unsigned* nodemax = (unsigned*)alloc((size_t)N_ * HEADS * sizeof(unsigned));
    float*    nodesum = (float*)alloc((size_t)N_ * HEADS * sizeof(float));
    float*    xbA     = (float*)alloc((size_t)N_ * HIDC * sizeof(float));
    float*    xbB     = (float*)alloc((size_t)N_ * HIDC * sizeof(float));

    // weight fragment conversion (tiny)
    for (int l = 0; l < 4; ++l) {
        int d_in_l = (l == 0) ? 40 : 136;
        int t0 = nkt1[l] * 4 * 512;
        cvt_w<<<(t0 + 255) / 256, 256, 0, stream>>>(gW[l][0], d_in_l, 64, nkt1[l], 4, wf[l][0]);
        int t1 = 2 * 4 * 512;
        cvt_w<<<(t1 + 255) / 256, 256, 0, stream>>>(gW[l][1], 64, 64, 2, 4, wf[l][1]);
        cvt_w<<<(t1 + 255) / 256, 256, 0, stream>>>(gW[l][2], 64, 64, 2, 4, wf[l][2]);
        int t3 = 2 * 1 * 512;
        cvt_w<<<(t3 + 255) / 256, 256, 0, stream>>>(gW[l][3], 64, 4, 2, 1, wf[l][3]);
    }

    const int blocksN64 = (N_ * HIDC + 255) / 256;
    const int blocksE   = (int)((E_ + 127) / 128);
    const float* xin = x;
    int d = 16;
    float* xout = xbA;

    for (int l = 0; l < 4; ++l) {
        init_layer<<<blocksN64, 256, 0, stream>>>(xout, nodesum, nodemax, N_);
        if (l == 0) {
            edge_mlp<2><<<blocksE, 256, 0, stream>>>(
                xin, d, 2 * d + 8, src, dst, eattr, (int)E_,
                wf[l][0], gB[l][0], wf[l][1], gB[l][1],
                wf[l][2], gB[l][2], wf[l][3], gB[l][3],
                hbuf, logit, nodemax);
        } else {
            edge_mlp<5><<<blocksE, 256, 0, stream>>>(
                xin, d, 2 * d + 8, src, dst, eattr, (int)E_,
                wf[l][0], gB[l][0], wf[l][1], gB[l][1],
                wf[l][2], gB[l][2], wf[l][3], gB[l][3],
                hbuf, logit, nodemax);
        }
        long E4 = E_ * HEADS;
        edge_exp<<<(int)((E4 + 255) / 256), 256, 0, stream>>>(dst, logit, nodemax, nodesum, E4);
        edge_w<<<(int)((E_ + 255) / 256), 256, 0, stream>>>(dst, logit, nodesum, wbuf, E_);
        long EC_ = E_ * HIDC;
        edge_aggr<<<(int)((EC_ + 255) / 256), 256, 0, stream>>>(dst, wbuf, hbuf, xout, EC_);
        node_gelu<<<blocksN64, 256, 0, stream>>>(xout, N_ * HIDC);
        xin = xout;
        d = 64;
        xout = (xout == xbA) ? xbB : xbA;
    }

    fc_kernel<<<(N_ + 255) / 256, 256, 0, stream>>>(
        xin, fcW1, fcb1, fcW2, fcb2, fcW3, fcb3, (float*)d_out, N_);
}
